// Model_61529701482572
// MI455X (gfx1250) — matrix operand order repaired and checked by reading, not compiled, not run on hardware
//
#include <hip/hip_runtime.h>

#define FD    41024    // 64 * (64*10 + 1)
#define KBASE 641      // 64*10 + 1
#define NB    1024
#define HALF  20512    // 32 * KBASE: king-square split point (= 641 col-tiles of 32)

typedef __attribute__((ext_vector_type(16))) _Float16 v16h;
typedef __attribute__((ext_vector_type(8)))  float    v8f;

// ---------------------------------------------------------------------------
// Kernel 1: tiled transpose of a column slice:
//   W[256][FD] cols [colOff, colOff+HALF)  ->  WT[HALF][256]
// 32x33 LDS tile; HALF = 32*641, rows = 32*8: exact tiling.
// ---------------------------------------------------------------------------
__global__ __launch_bounds__(256) void k_transpose(const float* __restrict__ in,
                                                   float* __restrict__ out,
                                                   int colOff) {
  __shared__ float tile[32][33];
  const int tx = threadIdx.x, ty = threadIdx.y;
  const int bx = blockIdx.x,  by = blockIdx.y;
#pragma unroll
  for (int i = 0; i < 4; ++i)
    tile[ty + i * 8][tx] =
        in[(size_t)(by * 32 + ty + i * 8) * FD + colOff + bx * 32 + tx];
  __syncthreads();
#pragma unroll
  for (int i = 0; i < 4; ++i)
    out[(size_t)(bx * 32 + ty + i * 8) * 256 + by * 32 + tx] = tile[tx][ty + i * 8];
}

// ---------------------------------------------------------------------------
// Kernel 2: fused HalfKP encode + sparse feature accumulation.
// One block per board, 512 threads:
//   threads [0,256)  -> h[b, 0:256]   = relu(b_my  + sum W_my [:, opp_idx])
//   threads [256,512)-> h[b, 256:512] = relu(b_opp + sum W_opp[:, my_idx ])
// (reference: feats = concat(opps, mys); my_in = opps, opp_in = mys)
// W access generalized: W[(idx - base)*sIdx + t*sT]
//   transposed half-matrix path: sIdx=256, sT=1, baseMy=HALF (opp_king>=32),
//   baseOpp=0 (my_king<32).  Fallback: strided original, bases=0.
// Output stored as f16 for the WMMA layer.
// ---------------------------------------------------------------------------
__global__ __launch_bounds__(512) void k_feat(const int* __restrict__ boards,
                                              const float* __restrict__ Wmy,
                                              const float* __restrict__ Wopp,
                                              const float* __restrict__ bmy,
                                              const float* __restrict__ bopp,
                                              _Float16* __restrict__ h,
                                              size_t sIdx, size_t sT,
                                              int baseMy, int baseOpp) {
  __shared__ int d[64];
  __shared__ int kmy, kopp;
  const int b   = blockIdx.x;
  const int tid = threadIdx.x;
  if (tid < 64) {
    int v = boards[b * 64 + tid];
    d[tid] = v;
    if (v == 0)  kmy  = tid;   // unique by construction
    if (v == 12) kopp = tid;   // unique by construction
  }
  __syncthreads();

  const int side = tid >> 8;      // 0: my_pos side, 1: opp_pos side
  const int t    = tid & 255;
  const float* __restrict__ W = side ? Wopp : Wmy;
  float acc = side ? bopp[t] : bmy[t];
  const int king = side ? kmy : kopp;   // my_pos uses opp_king, opp_pos uses my_king
  const int base = side ? baseOpp : baseMy;

  for (int sq = 0; sq < 64; ++sq) {
    const int v = d[sq];                 // wave-uniform
    bool act; int idx;
    if (side == 0) { act = (v >= 7) && (v <= 11); idx = king * KBASE + (11 - v) * 64 + sq; }
    else           { act = (v >= 1) && (v <= 5);  idx = king * KBASE + (v - 1)  * 64 + sq; }
    if (act) acc += W[(size_t)(idx - base) * sIdx + (size_t)t * sT];
  }
  h[(size_t)b * 512 + side * 256 + t] = (_Float16)fmaxf(acc, 0.0f);
}

// ---------------------------------------------------------------------------
// Kernel 3: layer 1 via V_WMMA_F32_16X16X32_F16.
// out1[1024,32] = relu(h[1024,512] @ W1.T + b1)
// One wave per (mtile, ntile): 64 M-tiles x 2 N-tiles = 128 waves, K loop = 16.
// Fragment layouts per CDNA5 ISA 7.12.2 (wave32); EXEC is full (no divergence).
// ---------------------------------------------------------------------------
__global__ __launch_bounds__(32) void k_layer1(const _Float16* __restrict__ h,
                                               const float* __restrict__ W1,
                                               const float* __restrict__ b1,
                                               float* __restrict__ out1) {
  const int lane  = threadIdx.x;
  const int mtile = blockIdx.x >> 1;
  const int ntile = blockIdx.x & 1;
  const int lo = lane & 15, hi = lane >> 4;
  const int row  = mtile * 16 + lo;   // A row (board) for this lane
  const int chan = ntile * 16 + lo;   // B col (output channel) for this lane

  v8f acc = {};
  for (int kb = 0; kb < 16; ++kb) {
    // A (16x32 f16): lane half 'hi' holds K = hi*8 + {0..7} and hi*8 + {16..23}
    const _Float16* ap = h + (size_t)row * 512 + kb * 32 + hi * 8;
    v16h a, bb;
#pragma unroll
    for (int e = 0; e < 8; ++e) { a[e] = ap[e]; a[e + 8] = ap[e + 16]; }
    // B (32x16): lane half 'hi' holds K = hi*8 + {0..7} and hi*8 + {16..23}, N = lo; W1 row-major [32][512]
    const float* bp = W1 + (size_t)chan * 512 + kb * 32 + hi * 8;
#pragma unroll
    for (int e = 0; e < 8; ++e) { bb[e] = (_Float16)bp[e]; bb[e + 8] = (_Float16)bp[e + 16]; }
    acc = __builtin_amdgcn_wmma_f32_16x16x32_f16(false, a, false, bb,
                                                 (short)0, acc, false, false);
  }
  const float bias = b1[chan];
#pragma unroll
  for (int r = 0; r < 8; ++r) {        // D: M = r + 8*hi, N = lo
    const int m = mtile * 16 + hi * 8 + r;
    out1[(size_t)m * 32 + chan] = fmaxf(acc[r] + bias, 0.0f);
  }
}

// ---------------------------------------------------------------------------
// Kernel 4: layers 2+3 in fp32, one wave32 per board.
// h2 = relu(h1 @ W2.T + b2); out = h2 @ W3.T + b3
// ---------------------------------------------------------------------------
__global__ __launch_bounds__(32) void k_head(const float* __restrict__ h1,
                                             const float* __restrict__ W2,
                                             const float* __restrict__ b2,
                                             const float* __restrict__ W3,
                                             const float* __restrict__ b3,
                                             float* __restrict__ out) {
  const int b = blockIdx.x, t = threadIdx.x;
  const float x = h1[(size_t)b * 32 + t];     // lane t holds h1[b][t]
  float acc = b2[t];
#pragma unroll
  for (int k = 0; k < 32; ++k)
    acc += W2[t * 32 + k] * __shfl(x, k, 32); // broadcast lane k
  const float h2 = fmaxf(acc, 0.0f);
  float p = W3[t] * h2;
#pragma unroll
  for (int off = 16; off > 0; off >>= 1)
    p += __shfl_down(p, off, 32);
  if (t == 0) out[b] = p + b3[0];
}

// ---------------------------------------------------------------------------
extern "C" void kernel_launch(void* const* d_in, const int* in_sizes, int n_in,
                              void* d_out, int out_size, void* d_ws, size_t ws_size,
                              hipStream_t stream) {
  (void)in_sizes; (void)n_in; (void)out_size;
  const int*   boards = (const int*)  d_in[0];
  const float* W_my   = (const float*)d_in[1];
  const float* b_my   = (const float*)d_in[2];
  const float* W_opp  = (const float*)d_in[3];
  const float* b_opp  = (const float*)d_in[4];
  const float* W1     = (const float*)d_in[5];
  const float* b1     = (const float*)d_in[6];
  const float* W2     = (const float*)d_in[7];
  const float* b2     = (const float*)d_in[8];
  const float* W3     = (const float*)d_in[9];
  const float* b3     = (const float*)d_in[10];
  float* out = (float*)d_out;

  const size_t wtBytes = (size_t)HALF * 256 * sizeof(float);   // 21,004,288 per half-matrix
  const size_t hBytes  = (size_t)NB * 512 * sizeof(_Float16);  // 1 MB
  const size_t h1Bytes = (size_t)NB * 32 * sizeof(float);      // 128 KB
  char* ws = (char*)d_ws;

  const bool useT = ws_size >= 2 * wtBytes + hBytes + h1Bytes; // ~43.2 MB
  const float *Wmy_p, *Wopp_p;
  size_t sIdx, sT;
  int baseMy, baseOpp;
  _Float16* hbuf;
  float* h1buf;

  if (useT) {
    // Only the king-reachable half of each matrix is ever gathered:
    //   W_my needs columns [HALF, FD)  (opp_king in [32,64))
    //   W_opp needs columns [0, HALF)  (my_king  in [0,32))
    float* WmyT  = (float*)ws;                 // holds W_my.T rows [HALF,FD) compacted
    float* WoppT = (float*)(ws + wtBytes);     // holds W_opp.T rows [0,HALF)
    hbuf  = (_Float16*)(ws + 2 * wtBytes);
    h1buf = (float*)(ws + 2 * wtBytes + hBytes);
    dim3 tb(32, 8), tg(HALF / 32, 256 / 32);   // 641 x 8 tiles
    k_transpose<<<tg, tb, 0, stream>>>(W_my,  WmyT,  HALF);
    k_transpose<<<tg, tb, 0, stream>>>(W_opp, WoppT, 0);
    Wmy_p = WmyT; Wopp_p = WoppT; sIdx = 256; sT = 1;
    baseMy = HALF; baseOpp = 0;
  } else {
    hbuf  = (_Float16*)ws;
    h1buf = (float*)(ws + hBytes);
    Wmy_p = W_my; Wopp_p = W_opp; sIdx = 1; sT = FD;   // strided fallback
    baseMy = 0; baseOpp = 0;
  }

  k_feat  <<<NB, 512, 0, stream>>>(boards, Wmy_p, Wopp_p, b_my, b_opp, hbuf,
                                   sIdx, sT, baseMy, baseOpp);
  k_layer1<<<(NB / 16) * 2, 32, 0, stream>>>(hbuf, W1, b1, h1buf);
  k_head  <<<NB, 32, 0, stream>>>(h1buf, W2, b2, W3, b3, out);
}
